// CellAnnotator_2645699854697
// MI455X (gfx1250) — matrix operand order repaired and checked by reading, not compiled, not run on hardware
//
#include <hip/hip_runtime.h>

// CDNA5 / gfx1250: wave32, WMMA 16x16x32 f16 -> f32
typedef __attribute__((ext_vector_type(16))) _Float16 v16h;
typedef __attribute__((ext_vector_type(8)))  float    v8f;

constexpr int kH     = 512;
constexpr int kW     = 512;
constexpr int kC     = 128;
constexpr int kROI   = 8;
constexpr int kPadLo = 3;    // XLA SAME for even kernel: lo=3, hi=4
constexpr int kCols  = 24;   // 23 valid halo columns + 1 zero-pad column

// One contiguous LDS block so "overflow" B reads (k>=24, multiplied by A==0)
// always land on finite fp16 data (xT slab or A-fragment halves), never NaN.
struct SMem {
    _Float16 xT[kROI * kC * kCols];  // [p][ch][col24] fp16 slab, 48 KB
    v16h     a[kROI * 32];           // banded-A WMMA fragments, 8 KB
    float    tot[16];                // cnts normalizer accumulators
    float    inv[16];                // 1/(tot + 1e-6), computed once per block
};

// Stage the 8-row x 23-col x 128-ch x-slab into LDS as fp16, transposed to
// [p][ch][col]. INTERIOR blocks skip all clamps/masks (block-uniform choice).
template <bool INTERIOR>
__device__ __forceinline__ void stage_slab(const float* __restrict__ x0,
                                           _Float16* __restrict__ xT,
                                           int i, int j0, int tid)
{
#pragma unroll
    for (int p = 0; p < kROI; ++p) {
        const int  row   = i + p - kPadLo;                  // wave-uniform
        const bool rowok = INTERIOR || ((row >= 0) && (row < kH));
        const float* xrow = x0 + (size_t)(rowok ? row : 0) * kW * kC;
        // 23 cols x 32 channel-quads = 736 float4 chunks per tap
        for (int t = tid; t < 23 * 32; t += 256) {
            const int col_k = t >> 5;                       // 0..22
            const int ch4   = (t & 31) << 2;                // 0,4,...,124
            const int col   = j0 - kPadLo + col_k;
            float4 v;
            if (INTERIOR) {
                v = *(const float4*)&xrow[(size_t)col * kC + ch4];
            } else {
                const int cc = min(max(col, 0), kW - 1);    // safe address
                v = make_float4(0.f, 0.f, 0.f, 0.f);
                if (rowok)                                  // uniform branch
                    v = *(const float4*)&xrow[(size_t)cc * kC + ch4];
                const bool ok = rowok && (col >= 0) && (col < kW);
                v.x = ok ? v.x : 0.f;
                v.y = ok ? v.y : 0.f;
                v.z = ok ? v.z : 0.f;
                v.w = ok ? v.w : 0.f;
            }
            _Float16* dst = &xT[((p * kC) + ch4) * kCols + col_k];
            dst[0 * kCols] = (_Float16)v.x;
            dst[1 * kCols] = (_Float16)v.y;
            dst[2 * kCols] = (_Float16)v.z;
            dst[3 * kCols] = (_Float16)v.w;
        }
    }
}

__global__ __launch_bounds__(256)
void att_wmma_kernel(const float* __restrict__ x0,    // [H,W,C]
                     const float* __restrict__ wts,   // [H,W,64]
                     const float* __restrict__ cnts,  // [H,W,1]
                     float* __restrict__ out)         // [H,W,C]
{
    __shared__ SMem sm;

    const int i    = blockIdx.y;        // output row
    const int j0   = blockIdx.x * 16;   // output column tile base
    const int tid  = threadIdx.x;
    const int wave = tid >> 5;          // 0..7 -> channel tile
    const int lane = tid & 31;
    const int c0   = wave * 16;
    const int u    = lane & 15;
    const int hi   = lane >> 4;

    // ---- phase 1: zero A fragments, totals, pad column; stage x slab ----
    {
        v16h z = {};
        sm.a[tid] = z;                  // 256 fragments, one per thread
    }
    if (tid < 16) sm.tot[tid] = 0.0f;
#pragma unroll
    for (int t = tid; t < kROI * kC; t += 256) {            // zero pad col 23
        const int p  = t >> 7;
        const int ch = t & (kC - 1);
        sm.xT[((p * kC) + ch) * kCols + (kCols - 1)] = (_Float16)0.0f;
    }

    const bool interior = (j0 >= kPadLo) && (j0 + 16 + 4 <= kW) &&
                          (i >= kPadLo) && (i + 4 < kH);     // block-uniform
    if (interior) stage_slab<true >(x0, sm.xT, i, j0, tid);
    else          stage_slab<false>(x0, sm.xT, i, j0, tid);

    // prefetch the slab row the next block-row will need (global_prefetch_b8)
    if (wave == 0) {
        const int rn = min(i + kROI - kPadLo + 1, kH - 1);
        __builtin_prefetch(x0 + ((size_t)rn * kW + j0) * kC + lane * 64, 0, 1);
    }
    __syncthreads();

    // ---- phase 2: scatter banded A + cnts normalizer, one thread per (p,u)
    if (tid < 128) {
        const int p  = tid >> 4;
        const int uu = tid & 15;
        const float* wp = wts + (((size_t)i * kW + (j0 + uu)) * 64 + p * kROI);
        float wq[8];
        *(float4*)&wq[0] = *(const float4*)&wp[0];          // 32B aligned
        *(float4*)&wq[4] = *(const float4*)&wp[4];

        _Float16* s_ah = (_Float16*)sm.a;
#pragma unroll
        for (int q = 0; q < kROI; ++q) {
            const int K  = uu + q;                          // 0..22
            const int ln = uu + 16 * (K >> 4);              // lane holding (u,K)
            const int e  = K & 15;                          // element in lane
            s_ah[(p * 32 + ln) * 16 + e] = (_Float16)wq[q];
        }

        const int row = i + p - kPadLo;                     // uniform per 16 thr
        if (row >= 0 && row < kH) {
            const float* crow = cnts + (size_t)row * kW;
            float part = 0.0f;
#pragma unroll
            for (int q = 0; q < kROI; ++q) {
                const int col = j0 + uu + q - kPadLo;
                const int cc  = min(max(col, 0), kW - 1);
                const float cv = crow[cc];
                part += ((col >= 0) && (col < kW)) ? cv * wq[q] : 0.0f;
            }
            atomicAdd(&sm.tot[uu], part);                   // ds_add_f32
        }
    }
    __syncthreads();

    // 16 reciprocals per BLOCK instead of 8 divides per THREAD; the WMMA
    // loop below hides the latency, barrier after it publishes sm.inv.
    if (tid < 16) sm.inv[tid] = 1.0f / (sm.tot[tid] + 1e-6f);

    // ---- phase 3: hot loop -- pure LDS b128 loads + WMMA, no masking ----
    v8f acc = {};
#pragma unroll
    for (int p = 0; p < kROI; ++p) {
        const v16h a = sm.a[p * 32 + lane];                 // 32B LDS load

        // B[k,n]: lane holds channel n=u; 16 contiguous cols starting hi*16
        const uint4* bp =
            (const uint4*)&sm.xT[((p * kC) + c0 + u) * kCols + hi * 16];
        union { v16h v; uint4 q[2]; } bb;
        bb.q[0] = bp[0];                                    // ds_load_b128
        bb.q[1] = bp[1];                                    // ds_load_b128

        // D = A x B + C : v_wmma_f32_16x16x32_f16
        acc = __builtin_amdgcn_wmma_f32_16x16x32_f16(
                  /*neg_a=*/false, a, /*neg_b=*/false, bb.v,
                  /*c_mod=*/(short)0, acc, /*reuse_a=*/false, /*reuse_b=*/false);
    }
    __syncthreads();                                        // sm.inv ready

    // ---- normalize + store. D layout: VGPR r -> M = r + hi*8, N = u ----
#pragma unroll
    for (int r = 0; r < 8; ++r) {
        const int m = r + hi * 8;                           // pixel within tile
        const float o = acc[r] * sm.inv[m];
        out[((size_t)i * kW + (j0 + m)) * kC + c0 + u] = o;
    }
}

extern "C" void kernel_launch(void* const* d_in, const int* in_sizes, int n_in,
                              void* d_out, int out_size, void* d_ws, size_t ws_size,
                              hipStream_t stream) {
    const float* x0   = (const float*)d_in[0];  // [512,512,128] f32
    const float* wts  = (const float*)d_in[1];  // [512,512,64]  f32
    const float* cnts = (const float*)d_in[2];  // [512,512,1]   f32
    float*       outp = (float*)d_out;          // [512,512,128] f32

    dim3 grid(kW / 16, kH);   // 32 x 512 blocks
    dim3 block(256);          // 8 waves: one 16-channel tile each
    att_wmma_kernel<<<grid, block, 0, stream>>>(x0, wts, cnts, outp);
}
